// LSTMAutoencoder_51427938402369
// MI455X (gfx1250) — matrix operand order repaired and checked by reading, not compiled, not run on hardware
//
#include <hip/hip_runtime.h>

#define BB 256
#define TT 128
#define II 512
#define HH 1024
#define EE 256

typedef __attribute__((ext_vector_type(16))) __bf16 v16bf;
typedef __attribute__((ext_vector_type(8)))  __bf16 v8bf;
typedef __attribute__((ext_vector_type(8)))  float  v8f;

#define WMMA_BF16(a, b, c) \
  __builtin_amdgcn_wmma_f32_16x16x32_bf16(false, (a), false, (b), (short)0, (c), false, false)

static __device__ __forceinline__ v16bf cat16(v8bf lo, v8bf hi) {
  return __builtin_shufflevector(lo, hi, 0,1,2,3,4,5,6,7,8,9,10,11,12,13,14,15);
}

// A fragment (16x32 bf16, wave32): lane L (hf = L/16) holds
// K = base+hf*16 .. +15 of its row (matching B) -> one contiguous
// 32-byte run of a row-major [M,K] matrix.
static __device__ __forceinline__ v16bf load_frag_a(const __bf16* rowk, int hf) {
  v8bf lo = *(const v8bf*)(rowk + hf * 16);
  v8bf hi = *(const v8bf*)(rowk + hf * 16 + 8);
  return cat16(lo, hi);
}

// B fragment (32x16 bf16): lane L holds column n=L%16, K = base+(L/16)*16 .. +15,
// contiguous in a row-major [N,K] weight -> one 32-byte run (two b128 loads).
static __device__ __forceinline__ v16bf load_frag_b(const __bf16* p) {
  v8bf lo = *(const v8bf*)(p);
  v8bf hi = *(const v8bf*)(p + 8);
  return cat16(lo, hi);
}

static __device__ __forceinline__ float sigmf(float x) {
  return 1.0f / (1.0f + __expf(-x));
}

// ---------------------------------------------------------------------------
// prep kernels
// ---------------------------------------------------------------------------
__global__ void f32_to_bf16_k(const float* __restrict__ in, __bf16* __restrict__ out, int n) {
  int i = blockIdx.x * 256 + threadIdx.x;
  int stride = gridDim.x * 256;
  for (; i < n; i += stride) out[i] = (__bf16)in[i];
}

__global__ void vec_add_k(const float* __restrict__ a, const float* __restrict__ b,
                          float* __restrict__ out, int n) {
  int i = blockIdx.x * 256 + threadIdx.x;
  int stride = gridDim.x * 256;
  for (; i < n; i += stride) out[i] = a[i] + b[i];
}

// Wcat[n][k] = k < II ? enc_Wih[n][k] : enc_Whh[n][k-II]   (bf16, [4H, I+H])
__global__ void build_encW_k(const float* __restrict__ Wih, const float* __restrict__ Whh,
                             __bf16* __restrict__ out, int n) {
  int i = blockIdx.x * 256 + threadIdx.x;
  int stride = gridDim.x * 256;
  for (; i < n; i += stride) {
    int row = i / (II + HH), k = i % (II + HH);
    float v = (k < II) ? Wih[(size_t)row * II + k] : Whh[(size_t)row * HH + (k - II)];
    out[i] = (__bf16)v;
  }
}

// Wrec[n][k] = dec_Wih[n][H+k] + dec_Whh[n][k]   (bf16, [4I, I])
__global__ void build_wrec_k(const float* __restrict__ dWih, const float* __restrict__ dWhh,
                             __bf16* __restrict__ out, int n) {
  int i = blockIdx.x * 256 + threadIdx.x;
  int stride = gridDim.x * 256;
  for (; i < n; i += stride) {
    int row = i / II, k = i % II;
    out[i] = (__bf16)(dWih[(size_t)row * (HH + II) + HH + k] + dWhh[(size_t)row * II + k]);
  }
}

// ---------------------------------------------------------------------------
// generic WMMA GEMM: out = act(A[M,K](bf16) x B[N,K]^T(bf16) + bias)
// one wave = one 16x64 strip; software-pipelined fragment loads
// ---------------------------------------------------------------------------
__global__ __launch_bounds__(256) void gemm_bias_act(
    const __bf16* __restrict__ A, int lda,
    const __bf16* __restrict__ Bw, int ldb,
    const float* __restrict__ bias,
    float* __restrict__ outF, __bf16* __restrict__ outB, int ldo,
    int N, int K, int doRelu)
{
  const int wave = threadIdx.x >> 5;
  const int lane = threadIdx.x & 31;
  const int hf   = lane >> 4;
  const int ln   = lane & 15;
  const int gw   = blockIdx.x * 8 + wave;
  const int nStrips = N >> 6;
  const int mt = gw / nStrips;
  const int ns = gw % nStrips;
  const int m0 = mt << 4;
  const int n0 = ns << 6;

  const v8f vz = {0.f,0.f,0.f,0.f,0.f,0.f,0.f,0.f};
  v8f acc0 = vz, acc1 = vz, acc2 = vz, acc3 = vz;

  const __bf16* arow = A + (size_t)(m0 + ln) * lda;
  const __bf16* w0 = Bw + (size_t)(n0 +  0 + ln) * ldb + hf * 16;
  const __bf16* w1 = Bw + (size_t)(n0 + 16 + ln) * ldb + hf * 16;
  const __bf16* w2 = Bw + (size_t)(n0 + 32 + ln) * ldb + hf * 16;
  const __bf16* w3 = Bw + (size_t)(n0 + 48 + ln) * ldb + hf * 16;

  v16bf a  = load_frag_a(arow, hf);
  v16bf b0 = load_frag_b(w0);
  v16bf b1 = load_frag_b(w1);
  v16bf b2 = load_frag_b(w2);
  v16bf b3 = load_frag_b(w3);

  for (int k = 0; k < K - 32; k += 32) {
    const int kn = k + 32;
    v16bf an  = load_frag_a(arow + kn, hf);
    v16bf b0n = load_frag_b(w0 + kn);
    v16bf b1n = load_frag_b(w1 + kn);
    v16bf b2n = load_frag_b(w2 + kn);
    v16bf b3n = load_frag_b(w3 + kn);
    acc0 = WMMA_BF16(a, b0, acc0);
    acc1 = WMMA_BF16(a, b1, acc1);
    acc2 = WMMA_BF16(a, b2, acc2);
    acc3 = WMMA_BF16(a, b3, acc3);
    a = an; b0 = b0n; b1 = b1n; b2 = b2n; b3 = b3n;
  }
  acc0 = WMMA_BF16(a, b0, acc0);
  acc1 = WMMA_BF16(a, b1, acc1);
  acc2 = WMMA_BF16(a, b2, acc2);
  acc3 = WMMA_BF16(a, b3, acc3);

  v8f accs[4] = {acc0, acc1, acc2, acc3};
#pragma unroll
  for (int q = 0; q < 4; ++q) {
    int n = n0 + q * 16 + ln;
    float bv = bias ? bias[n] : 0.0f;
#pragma unroll
    for (int v = 0; v < 8; ++v) {
      int m = m0 + hf * 8 + v;
      float val = accs[q][v] + bv;
      if (doRelu) val = fmaxf(val, 0.0f);
      if (outF) outF[(size_t)m * ldo + n] = val;
      if (outB) outB[(size_t)m * ldo + n] = (__bf16)val;
    }
  }
}

// ---------------------------------------------------------------------------
// encoder step: pre = [x_t | h] @ Wcat^T + bsum ; fused LSTM gates
// grid 64 blocks: 4 m-blocks (64 rows) x 16 j-blocks (64 hidden cols)
// wave w -> (gate = w>>1, mi = w&1): 32 rows x 64 cols of gate `gate`
// (2 A-frags x 4 B-frags -> 8 WMMA per k-step, double-buffered)
// ---------------------------------------------------------------------------
__global__ __launch_bounds__(256) void enc_step_kernel(
    const __bf16* __restrict__ xbf,   // [B,T,I] bf16
    const __bf16* __restrict__ hin,   // [B,H] bf16
    __bf16* __restrict__ hout,        // [B,H] bf16
    const __bf16* __restrict__ W,     // [4H, I+H] bf16
    const float* __restrict__ bsum,   // [4H]
    float* __restrict__ c,            // [B,H] f32 (persistent)
    int t)
{
  __shared__ float lds[4][64][64];
  const int wave = threadIdx.x >> 5;
  const int lane = threadIdx.x & 31;
  const int hf   = lane >> 4;
  const int ln   = lane & 15;
  const int mb   = blockIdx.x >> 4;
  const int jb   = blockIdx.x & 15;
  const int mi   = wave & 1;
  const int gate = wave >> 1;
  const int m0   = mb * 64 + mi * 32;
  const int am0  = m0 + ln;
  const int am1  = m0 + 16 + ln;
  const int n0   = gate * HH + jb * 64;
  const int ldw  = II + HH;
  const int KT   = II + HH;

  const v8f vz = {0.f,0.f,0.f,0.f,0.f,0.f,0.f,0.f};
  v8f acc00 = vz, acc01 = vz, acc02 = vz, acc03 = vz;
  v8f acc10 = vz, acc11 = vz, acc12 = vz, acc13 = vz;

  const __bf16* xr0 = xbf + ((size_t)am0 * TT + t) * II;
  const __bf16* xr1 = xbf + ((size_t)am1 * TT + t) * II;
  const __bf16* hr0 = hin + (size_t)am0 * HH - II;   // hr + k valid for k >= II
  const __bf16* hr1 = hin + (size_t)am1 * HH - II;
  const __bf16* w0 = W + (size_t)(n0 +  0 + ln) * ldw + hf * 16;
  const __bf16* w1 = W + (size_t)(n0 + 16 + ln) * ldw + hf * 16;
  const __bf16* w2 = W + (size_t)(n0 + 32 + ln) * ldw + hf * 16;
  const __bf16* w3 = W + (size_t)(n0 + 48 + ln) * ldw + hf * 16;

  v16bf a0 = load_frag_a(xr0, hf);
  v16bf a1 = load_frag_a(xr1, hf);
  v16bf b0 = load_frag_b(w0);
  v16bf b1 = load_frag_b(w1);
  v16bf b2 = load_frag_b(w2);
  v16bf b3 = load_frag_b(w3);

  for (int k = 0; k < KT - 32; k += 32) {
    const int kn = k + 32;
    const __bf16* ar0 = (kn < II) ? (xr0 + kn) : (hr0 + kn);
    const __bf16* ar1 = (kn < II) ? (xr1 + kn) : (hr1 + kn);
    v16bf a0n = load_frag_a(ar0, hf);
    v16bf a1n = load_frag_a(ar1, hf);
    v16bf b0n = load_frag_b(w0 + kn);
    v16bf b1n = load_frag_b(w1 + kn);
    v16bf b2n = load_frag_b(w2 + kn);
    v16bf b3n = load_frag_b(w3 + kn);
    acc00 = WMMA_BF16(a0, b0, acc00);
    acc10 = WMMA_BF16(a1, b0, acc10);
    acc01 = WMMA_BF16(a0, b1, acc01);
    acc11 = WMMA_BF16(a1, b1, acc11);
    acc02 = WMMA_BF16(a0, b2, acc02);
    acc12 = WMMA_BF16(a1, b2, acc12);
    acc03 = WMMA_BF16(a0, b3, acc03);
    acc13 = WMMA_BF16(a1, b3, acc13);
    a0 = a0n; a1 = a1n; b0 = b0n; b1 = b1n; b2 = b2n; b3 = b3n;
  }
  acc00 = WMMA_BF16(a0, b0, acc00);
  acc10 = WMMA_BF16(a1, b0, acc10);
  acc01 = WMMA_BF16(a0, b1, acc01);
  acc11 = WMMA_BF16(a1, b1, acc11);
  acc02 = WMMA_BF16(a0, b2, acc02);
  acc12 = WMMA_BF16(a1, b2, acc12);
  acc03 = WMMA_BF16(a0, b3, acc03);
  acc13 = WMMA_BF16(a1, b3, acc13);

  {
    const int rb0 = mi * 32 + hf * 8;
    const int rb1 = mi * 32 + 16 + hf * 8;
#pragma unroll
    for (int v = 0; v < 8; ++v) {
      lds[gate][rb0 + v][ 0 + ln] = acc00[v];
      lds[gate][rb0 + v][16 + ln] = acc01[v];
      lds[gate][rb0 + v][32 + ln] = acc02[v];
      lds[gate][rb0 + v][48 + ln] = acc03[v];
      lds[gate][rb1 + v][ 0 + ln] = acc10[v];
      lds[gate][rb1 + v][16 + ln] = acc11[v];
      lds[gate][rb1 + v][32 + ln] = acc12[v];
      lds[gate][rb1 + v][48 + ln] = acc13[v];
    }
  }
  __syncthreads();

#pragma unroll
  for (int e = 0; e < 16; ++e) {
    int f   = threadIdx.x + 256 * e;   // 0..4095 over 64x64 tile
    int row = f >> 6, col = f & 63;
    int b = mb * 64 + row;
    int j = jb * 64 + col;
    float ig = sigmf(lds[0][row][col] + bsum[j]);
    float fg = sigmf(lds[1][row][col] + bsum[HH + j]);
    float gg = tanhf(lds[2][row][col] + bsum[2 * HH + j]);
    float og = sigmf(lds[3][row][col] + bsum[3 * HH + j]);
    size_t idx = (size_t)b * HH + j;
    float cn = fg * c[idx] + ig * gg;
    c[idx] = cn;
    hout[idx] = (__bf16)(og * tanhf(cn));
  }
}

// ---------------------------------------------------------------------------
// decoder step: pre = fixed + h @ Wrec^T ; fused gates; h -> out[b, T-1-s, :]
// grid 32 blocks: 4 m-blocks (64 rows) x 8 j-blocks (64 of 512 cols)
// ---------------------------------------------------------------------------
__global__ __launch_bounds__(256) void dec_step_kernel(
    const __bf16* __restrict__ hin,    // [B,I] bf16
    __bf16* __restrict__ hout,         // [B,I] bf16
    const __bf16* __restrict__ Wrec,   // [4I, I] bf16
    const float* __restrict__ fixedP,  // [B,4I] f32 (includes biases)
    float* __restrict__ c,             // [B,I] f32
    float* __restrict__ out,           // [B,T,I] f32
    int s)
{
  __shared__ float lds[4][64][64];
  const int wave = threadIdx.x >> 5;
  const int lane = threadIdx.x & 31;
  const int hf   = lane >> 4;
  const int ln   = lane & 15;
  const int mb   = blockIdx.x >> 3;
  const int jb   = blockIdx.x & 7;
  const int mi   = wave & 1;
  const int gate = wave >> 1;
  const int m0   = mb * 64 + mi * 32;
  const int am0  = m0 + ln;
  const int am1  = m0 + 16 + ln;
  const int n0   = gate * II + jb * 64;

  const v8f vz = {0.f,0.f,0.f,0.f,0.f,0.f,0.f,0.f};
  v8f acc00 = vz, acc01 = vz, acc02 = vz, acc03 = vz;
  v8f acc10 = vz, acc11 = vz, acc12 = vz, acc13 = vz;

  const __bf16* hr0 = hin + (size_t)am0 * II;
  const __bf16* hr1 = hin + (size_t)am1 * II;
  const __bf16* w0 = Wrec + (size_t)(n0 +  0 + ln) * II + hf * 16;
  const __bf16* w1 = Wrec + (size_t)(n0 + 16 + ln) * II + hf * 16;
  const __bf16* w2 = Wrec + (size_t)(n0 + 32 + ln) * II + hf * 16;
  const __bf16* w3 = Wrec + (size_t)(n0 + 48 + ln) * II + hf * 16;

  v16bf a0 = load_frag_a(hr0, hf);
  v16bf a1 = load_frag_a(hr1, hf);
  v16bf b0 = load_frag_b(w0);
  v16bf b1 = load_frag_b(w1);
  v16bf b2 = load_frag_b(w2);
  v16bf b3 = load_frag_b(w3);

  for (int k = 0; k < II - 32; k += 32) {
    const int kn = k + 32;
    v16bf a0n = load_frag_a(hr0 + kn, hf);
    v16bf a1n = load_frag_a(hr1 + kn, hf);
    v16bf b0n = load_frag_b(w0 + kn);
    v16bf b1n = load_frag_b(w1 + kn);
    v16bf b2n = load_frag_b(w2 + kn);
    v16bf b3n = load_frag_b(w3 + kn);
    acc00 = WMMA_BF16(a0, b0, acc00);
    acc10 = WMMA_BF16(a1, b0, acc10);
    acc01 = WMMA_BF16(a0, b1, acc01);
    acc11 = WMMA_BF16(a1, b1, acc11);
    acc02 = WMMA_BF16(a0, b2, acc02);
    acc12 = WMMA_BF16(a1, b2, acc12);
    acc03 = WMMA_BF16(a0, b3, acc03);
    acc13 = WMMA_BF16(a1, b3, acc13);
    a0 = a0n; a1 = a1n; b0 = b0n; b1 = b1n; b2 = b2n; b3 = b3n;
  }
  acc00 = WMMA_BF16(a0, b0, acc00);
  acc10 = WMMA_BF16(a1, b0, acc10);
  acc01 = WMMA_BF16(a0, b1, acc01);
  acc11 = WMMA_BF16(a1, b1, acc11);
  acc02 = WMMA_BF16(a0, b2, acc02);
  acc12 = WMMA_BF16(a1, b2, acc12);
  acc03 = WMMA_BF16(a0, b3, acc03);
  acc13 = WMMA_BF16(a1, b3, acc13);

  {
    const int rb0 = mi * 32 + hf * 8;
    const int rb1 = mi * 32 + 16 + hf * 8;
#pragma unroll
    for (int v = 0; v < 8; ++v) {
      lds[gate][rb0 + v][ 0 + ln] = acc00[v];
      lds[gate][rb0 + v][16 + ln] = acc01[v];
      lds[gate][rb0 + v][32 + ln] = acc02[v];
      lds[gate][rb0 + v][48 + ln] = acc03[v];
      lds[gate][rb1 + v][ 0 + ln] = acc10[v];
      lds[gate][rb1 + v][16 + ln] = acc11[v];
      lds[gate][rb1 + v][32 + ln] = acc12[v];
      lds[gate][rb1 + v][48 + ln] = acc13[v];
    }
  }
  __syncthreads();

#pragma unroll
  for (int e = 0; e < 16; ++e) {
    int f   = threadIdx.x + 256 * e;   // 0..4095 over 64x64 tile
    int row = f >> 6, col = f & 63;
    int b = mb * 64 + row;
    int j = jb * 64 + col;
    size_t fb = (size_t)b * (4 * II);
    float ig = sigmf(lds[0][row][col] + fixedP[fb + j]);
    float fg = sigmf(lds[1][row][col] + fixedP[fb + II + j]);
    float gg = tanhf(lds[2][row][col] + fixedP[fb + 2 * II + j]);
    float og = sigmf(lds[3][row][col] + fixedP[fb + 3 * II + j]);
    size_t idx = (size_t)b * II + j;
    float cn = fg * c[idx] + ig * gg;
    c[idx] = cn;
    float h = og * tanhf(cn);
    out[((size_t)b * TT + (TT - 1 - s)) * II + j] = h;   // reversed time
    hout[idx] = (__bf16)h;
  }
}

// ---------------------------------------------------------------------------
extern "C" void kernel_launch(void* const* d_in, const int* in_sizes, int n_in,
                              void* d_out, int out_size, void* d_ws, size_t ws_size,
                              hipStream_t stream) {
  (void)in_sizes; (void)n_in; (void)out_size; (void)ws_size;
  const float* x      = (const float*)d_in[0];
  const float* encWih = (const float*)d_in[1];
  const float* encWhh = (const float*)d_in[2];
  const float* encbih = (const float*)d_in[3];
  const float* encbhh = (const float*)d_in[4];
  const float* encfcW = (const float*)d_in[5];
  const float* encfcb = (const float*)d_in[6];
  const float* decfcW = (const float*)d_in[7];
  const float* decfcb = (const float*)d_in[8];
  const float* decWih = (const float*)d_in[9];
  const float* decWhh = (const float*)d_in[10];
  const float* decbih = (const float*)d_in[11];
  const float* decbhh = (const float*)d_in[12];
  float* out = (float*)d_out;

  char* ws = (char*)d_ws;
  size_t off = 0;
  auto take = [&](size_t bytes) -> char* {
    off = (off + 255) & ~(size_t)255;
    char* p = ws + off;
    off += bytes;
    return p;
  };

  __bf16* xbf    = (__bf16*)take((size_t)BB * TT * II * 2);        // 32 MB
  __bf16* Wenc   = (__bf16*)take((size_t)4 * HH * (II + HH) * 2);  // 12 MB
  float*  bsumE  = (float*) take((size_t)4 * HH * 4);
  __bf16* dWihB  = (__bf16*)take((size_t)4 * II * (HH + II) * 2);  // 6 MB
  __bf16* WrecB  = (__bf16*)take((size_t)4 * II * II * 2);         // 2 MB
  float*  bsumD  = (float*) take((size_t)4 * II * 4);
  __bf16* eFCW   = (__bf16*)take((size_t)EE * HH * 2);
  __bf16* dFCW   = (__bf16*)take((size_t)HH * EE * 2);
  __bf16* hEnc   = (__bf16*)take((size_t)2 * BB * HH * 2);         // double buffer
  float*  cEnc   = (float*) take((size_t)BB * HH * 4);
  __bf16* cEncB  = (__bf16*)take((size_t)BB * HH * 2);
  __bf16* embdB  = (__bf16*)take((size_t)BB * EE * 2);
  __bf16* dfhB   = (__bf16*)take((size_t)BB * HH * 2);
  float*  fixedP = (float*) take((size_t)BB * 4 * II * 4);
  __bf16* hDec   = (__bf16*)take((size_t)2 * BB * II * 2);         // double buffer
  float*  cDec   = (float*) take((size_t)BB * II * 4);

  auto gs = [](size_t n) { size_t b = (n + 255) / 256; return (int)(b > 2048 ? 2048 : b); };

  // --- one-time prep: bf16 weight/activation copies, bias sums ---
  f32_to_bf16_k<<<gs((size_t)BB * TT * II), 256, 0, stream>>>(x, xbf, BB * TT * II);
  build_encW_k<<<gs((size_t)4 * HH * (II + HH)), 256, 0, stream>>>(encWih, encWhh, Wenc, 4 * HH * (II + HH));
  vec_add_k<<<gs(4 * HH), 256, 0, stream>>>(encbih, encbhh, bsumE, 4 * HH);
  f32_to_bf16_k<<<gs((size_t)4 * II * (HH + II)), 256, 0, stream>>>(decWih, dWihB, 4 * II * (HH + II));
  build_wrec_k<<<gs((size_t)4 * II * II), 256, 0, stream>>>(decWih, decWhh, WrecB, 4 * II * II);
  vec_add_k<<<gs(4 * II), 256, 0, stream>>>(decbih, decbhh, bsumD, 4 * II);
  f32_to_bf16_k<<<gs((size_t)EE * HH), 256, 0, stream>>>(encfcW, eFCW, EE * HH);
  f32_to_bf16_k<<<gs((size_t)HH * EE), 256, 0, stream>>>(decfcW, dFCW, HH * EE);

  hipMemsetAsync(hEnc, 0, (size_t)2 * BB * HH * 2, stream);
  hipMemsetAsync(cEnc, 0, (size_t)BB * HH * 4, stream);
  hipMemsetAsync(hDec, 0, (size_t)2 * BB * II * 2, stream);
  hipMemsetAsync(cDec, 0, (size_t)BB * II * 4, stream);

  // --- encoder recurrence: 128 fused GEMM+gate steps ---
  for (int t = 0; t < TT; ++t) {
    enc_step_kernel<<<64, 256, 0, stream>>>(
        xbf, hEnc + (size_t)(t & 1) * BB * HH, hEnc + (size_t)((t + 1) & 1) * BB * HH,
        Wenc, bsumE, cEnc, t);
  }
  f32_to_bf16_k<<<gs((size_t)BB * HH), 256, 0, stream>>>(cEnc, cEncB, BB * HH);

  // embd = relu(c_final @ encfcW^T + b)   M=256,N=256,K=1024 (f32 -> d_out tail, bf16 copy)
  gemm_bias_act<<<(16 * (EE / 64)) / 8, 256, 0, stream>>>(
      cEncB, HH, eFCW, HH, encfcb, out + (size_t)BB * TT * II, embdB, EE, EE, HH, 1);
  // dec_first_hidden = relu(embd @ decfcW^T + b)   M=256,N=1024,K=256
  gemm_bias_act<<<(16 * (HH / 64)) / 8, 256, 0, stream>>>(
      embdB, EE, dFCW, EE, decfcb, nullptr, dfhB, HH, HH, EE, 1);
  // fixed = dfh @ dec_Wih[:, :H]^T + (bih+bhh)   M=256,N=2048,K=1024 (ldb spans full row)
  gemm_bias_act<<<(16 * ((4 * II) / 64)) / 8, 256, 0, stream>>>(
      dfhB, HH, dWihB, HH + II, bsumD, fixedP, nullptr, 4 * II, 4 * II, HH, 0);

  // --- decoder recurrence: 128 fused GEMM+gate steps, writes reversed output ---
  for (int s = 0; s < TT; ++s) {
    dec_step_kernel<<<32, 256, 0, stream>>>(
        hDec + (size_t)(s & 1) * BB * II, hDec + (size_t)((s + 1) & 1) * BB * II,
        WrecB, fixedP, cDec, out, s);
  }
}
